// NeuralCRFTagger_60086592471703
// MI455X (gfx1250) — matrix operand order repaired and checked by reading, not compiled, not run on hardware
//
#include <hip/hip_runtime.h>
#include <hip/hip_bf16.h>

typedef unsigned int u32;
typedef __attribute__((ext_vector_type(16))) __bf16 v16bf;
typedef __attribute__((ext_vector_type(8)))  float  v8f;

union Frag {
  v16bf v;
  u32   u[8];
  uint4 q[2];
};

__device__ __forceinline__ u32 pack_bf16(float lo, float hi) {
  u32 a = __float_as_uint(lo);
  u32 b = __float_as_uint(hi);
  a = (a + 0x7FFFu + ((a >> 16) & 1u)) >> 16;   // round-to-nearest-even
  b = (b + 0x7FFFu + ((b >> 16) & 1u)) >> 16;
  return (a & 0xFFFFu) | (b << 16);
}

#define BATCH   256
#define S_LEN   512
#define EMB_D   256
#define HID_D   512
#define NTAG    48

// ---- LDS layout (dword offsets) ----
#define A1_STRIDE 132          // 64 rows x 132 dw (256 bf16 padded); reused as EM 64x68 f32
#define LDS_A1    0
#define WB_STRIDE 20           // transposed W slice: 512 n x 20 dw (16 kp padded)
#define LDS_WB    8448
#define H_STRIDE  260          // 64 rows x 260 dw (512 bf16 padded)
#define LDS_H     18688
#define LDS_BF    35328        // 512 f32 b_feat
#define LDS_BE    35840        // 64 f32 b_emit (padded)
#define LDS_TOTAL_DW 35904     // 143616 bytes

// ============================================================
// Kernel 0: pack weights to bf16 in WMMA-friendly layouts
//   WfeatPT: [n=512][kp=128] dwords, dword = (W[2kp][n], W[2kp+1][n])
//            (kp contiguous -> async b128 units are contiguous in global)
//   WemitPT: [n=64][kp=256]  dwords (transposed, tags padded 48->64)
// ============================================================
__global__ __launch_bounds__(256) void crf_prep(
    const float* __restrict__ Wf, const float* __restrict__ We,
    u32* __restrict__ WfeatPT, u32* __restrict__ WemitPT)
{
  int idx = blockIdx.x * 256 + threadIdx.x;
  if (idx < 512 * 128) {
    int n = idx >> 7, kp = idx & 127;
    WfeatPT[idx] = pack_bf16(Wf[(2 * kp) * HID_D + n], Wf[(2 * kp + 1) * HID_D + n]);
  }
  if (idx < 64 * 256) {
    int n = idx >> 8, kp = idx & 255;
    u32 v = 0u;
    if (n < NTAG)
      v = pack_bf16(We[(2 * kp) * NTAG + n], We[(2 * kp + 1) * NTAG + n]);
    WemitPT[n * 256 + kp] = v;
  }
}

// ============================================================
// Kernel 1: fused gather -> GEMM1(tanh) -> GEMM2 -> token reduce
// 1 block = 256 threads (8 waves) handles 64 tokens (4 M-tiles)
// ============================================================
__global__ __launch_bounds__(256) void crf_fused(
    const int*   __restrict__ seq,   const float* __restrict__ emb,
    const float* __restrict__ bfeat, const float* __restrict__ bemit,
    const int*   __restrict__ targets,
    const u32*   __restrict__ WfeatPT, const u32* __restrict__ WemitPT,
    float* __restrict__ esum_out, float* __restrict__ emgold_out)
{
  extern __shared__ u32 lds[];
  u32*   A1 = lds + LDS_A1;
  u32*   WB = lds + LDS_WB;
  u32*   H  = lds + LDS_H;
  float* BF = (float*)(lds + LDS_BF);
  float* BE = (float*)(lds + LDS_BE);

  const int tid  = threadIdx.x;
  const int lane = tid & 31;
  const int wave = tid >> 5;
  const int col  = lane & 15;
  const int half = lane >> 4;
  const int tokbase = blockIdx.x * 64;

  // biases to LDS
  for (int i = tid; i < HID_D; i += 256) BF[i] = bfeat[i];
  if (tid < 64) BE[tid] = (tid < NTAG) ? bemit[tid] : 0.0f;

  // gather embeddings -> bf16 A tile (64 x 256); one wave streams one row
  for (int u = tid; u < 64 * 32; u += 256) {
    int row = u >> 5, seg = u & 31;
    int token = seq[tokbase + row];
    const float4* src = (const float4*)(emb + (size_t)token * EMB_D + seg * 8);
    float4 x0 = src[0];
    float4 x1 = src[1];
    u32* dst = A1 + row * A1_STRIDE + seg * 4;
    dst[0] = pack_bf16(x0.x, x0.y);
    dst[1] = pack_bf16(x0.z, x0.w);
    dst[2] = pack_bf16(x1.x, x1.y);
    dst[3] = pack_bf16(x1.z, x1.w);
  }

  const v8f vzero = {0.f, 0.f, 0.f, 0.f, 0.f, 0.f, 0.f, 0.f};

  // ---- GEMM1: h = tanh(x @ W_feat + b_feat), block tile 64x512 ----
  const int mt = wave & 3;   // M tile
  const int nh = wave >> 2;  // N half: 16 N-tiles each
  v8f acc[16];
#pragma unroll
  for (int i = 0; i < 16; ++i) acc[i] = vzero;

  // per-thread async-copy unit: covers (n, kp-quad); fixed across kt
  const int au_n[8] = { (tid + 0*256) >> 2, (tid + 1*256) >> 2, (tid + 2*256) >> 2,
                        (tid + 3*256) >> 2, (tid + 4*256) >> 2, (tid + 5*256) >> 2,
                        (tid + 6*256) >> 2, (tid + 7*256) >> 2 };
  const int au_q = tid & 3;

  for (int kt = 0; kt < 8; ++kt) {           // K = 256 = 8 x 32
    __syncthreads();                         // WB consumable again
    // async copy: global (contiguous b128) -> LDS (padded transpose layout)
#pragma unroll
    for (int k = 0; k < 8; ++k) {
      int n = au_n[k];
      u32 ldsaddr = (u32)((LDS_WB + n * WB_STRIDE + au_q * 4) * 4);
      u32 goff    = (u32)((n * 128 + au_q * 4) * 4 + kt * 64);
      asm volatile("global_load_async_to_lds_b128 %0, %1, %2"
                   :: "v"(ldsaddr), "v"(goff), "s"(WfeatPT)
                   : "memory");
    }
    asm volatile("s_wait_asynccnt 0x0" ::: "memory");
    __syncthreads();

    Frag a;                                  // A-frag: lane=row, K packed
    const u32* abase = A1 + (mt * 16 + col) * A1_STRIDE + kt * 16 + 8 * half;
    a.q[0] = *(const uint4*)abase;
    a.q[1] = *(const uint4*)(abase + 4);

#pragma unroll
    for (int nt = 0; nt < 16; ++nt) {
      int n = (nh * 16 + nt) * 16 + col;     // B-frag: lane=col, kp=8*half+j
      Frag b;
      const u32* bbase = WB + n * WB_STRIDE + 8 * half;
      b.q[0] = *(const uint4*)bbase;
      b.q[1] = *(const uint4*)(bbase + 4);
      acc[nt] = __builtin_amdgcn_wmma_f32_16x16x32_bf16(
          false, a.v, false, b.v, (short)0, acc[nt], false, false);
    }
  }
  __syncthreads();

  // epilogue 1: bias + tanh -> H (bf16). C/D layout: row = 8*half+v, col = lane&15
  unsigned short* H16 = (unsigned short*)H;
#pragma unroll
  for (int nt = 0; nt < 16; ++nt) {
    int n = (nh * 16 + nt) * 16 + col;
    float bias = BF[n];
#pragma unroll
    for (int v = 0; v < 8; ++v) {
      int row = mt * 16 + 8 * half + v;
      float hv = tanhf(acc[nt][v] + bias);
      H16[row * (H_STRIDE * 2) + n] = (unsigned short)pack_bf16(hv, 0.f);
    }
  }
  __syncthreads();

  // ---- GEMM2: em = h @ W_emit + b_emit, block tile 64x64 ----
  const int m2  = wave & 3;
  const int np2 = wave >> 2;   // N-tile pair: {0,1} or {2,3}
  v8f acc2[2];
  acc2[0] = vzero; acc2[1] = vzero;

  for (int kt = 0; kt < 16; ++kt) {          // K = 512 = 16 x 32
    Frag a;
    const u32* abase = H + (m2 * 16 + col) * H_STRIDE + kt * 16 + 8 * half;
    a.q[0] = *(const uint4*)abase;
    a.q[1] = *(const uint4*)(abase + 4);
#pragma unroll
    for (int j = 0; j < 2; ++j) {
      int ntile = np2 * 2 + j;
      Frag b;                                // W_emit is L2-resident: stream direct
      const uint4* bb = (const uint4*)(WemitPT + (ntile * 16 + col) * 256 + kt * 16 + 8 * half);
      b.q[0] = bb[0];
      b.q[1] = bb[1];
      acc2[j] = __builtin_amdgcn_wmma_f32_16x16x32_bf16(
          false, a.v, false, b.v, (short)0, acc2[j], false, false);
    }
  }
  __syncthreads();                           // A1 region now reused as EM

  float* EM = (float*)A1;                    // 64 x 68 f32
#pragma unroll
  for (int j = 0; j < 2; ++j) {
    int nb = (np2 * 2 + j) * 16 + col;
    float bias = BE[nb];
#pragma unroll
    for (int v = 0; v < 8; ++v) {
      int row = m2 * 16 + 8 * half + v;
      EM[row * 68 + nb] = acc2[j][v] + bias;
    }
  }
  __syncthreads();

  // per-token reduce: esum over 48 tags + gold emission, with pad mask
  if (tid < 64) {
    int g = tokbase + tid;
    int token = seq[g];
    float es = 0.f, eg = 0.f;
    if (token != 0) {
      const float* er = EM + tid * 68;
#pragma unroll 8
      for (int c = 0; c < NTAG; ++c) es += er[c];
      int tg = targets[g];
      if (tg < 0) tg = 0;
      eg = er[tg];
    }
    esum_out[g]   = es;
    emgold_out[g] = eg;
  }
}

// ============================================================
// Kernel 2: scalar recurrence per batch row + gold score
//   asum_t = 48*asum_{t-1} + (pad?0:Tsum) + 48*esum_t
//   log_Z  = asum_S + sum(stop);  gold = start[t0] + Σem_gold + Σtrans
// ============================================================
__global__ __launch_bounds__(256) void crf_final(
    const int* __restrict__ seq, const int* __restrict__ targets,
    const float* __restrict__ trans, const float* __restrict__ start_t,
    const float* __restrict__ stop_t, const float* __restrict__ esum,
    const float* __restrict__ emgold, float* __restrict__ out)
{
  __shared__ float TR[NTAG * NTAG];
  __shared__ float part[256];
  __shared__ float sc[3];
  const int tid = threadIdx.x;

  float p = 0.f;
  for (int i = tid; i < NTAG * NTAG; i += 256) {
    float t = trans[i];
    TR[i] = t;
    p += t;
  }
  part[tid] = p;
  __syncthreads();
  if (tid == 0) {
    float T = 0.f;
    for (int i = 0; i < 256; ++i) T += part[i];
    float ss = 0.f, ps = 0.f;
    for (int i = 0; i < NTAG; ++i) { ss += start_t[i]; ps += stop_t[i]; }
    sc[0] = T; sc[1] = ss; sc[2] = ps;
  }
  __syncthreads();

  const int b = tid;                 // one thread per batch row (B=256)
  const float Tsum = sc[0];
  float asum = sc[1];
  const float* es = esum   + b * S_LEN;
  const float* eg = emgold + b * S_LEN;
  const int*   sq = seq     + b * S_LEN;
  const int*   tg = targets + b * S_LEN;

  int prev = tg[0]; if (prev < 0) prev = 0;
  float gold = start_t[prev];
  float egsum = 0.f;
  for (int t = 0; t < S_LEN; ++t) {
    float c = (sq[t] == 0 ? 0.f : Tsum) + 48.f * es[t];
    asum = 48.f * asum + c;
    egsum += eg[t];
    if (t >= 1) {
      int cur = tg[t]; if (cur < 0) cur = 0;
      gold += TR[cur * NTAG + prev];
      prev = cur;
    }
  }
  out[b]         = asum + sc[2];     // log_Z
  out[BATCH + b] = gold + egsum;     // gold score
}

// ============================================================
extern "C" void kernel_launch(void* const* d_in, const int* in_sizes, int n_in,
                              void* d_out, int out_size, void* d_ws, size_t ws_size,
                              hipStream_t stream) {
  (void)in_sizes; (void)n_in; (void)out_size; (void)ws_size;
  const int*   seq     = (const int*)  d_in[0];
  // d_in[1] = length (unused by reference)
  const int*   targets = (const int*)  d_in[2];
  const float* emb     = (const float*)d_in[3];
  const float* Wf      = (const float*)d_in[4];
  const float* bf      = (const float*)d_in[5];
  const float* We      = (const float*)d_in[6];
  const float* be      = (const float*)d_in[7];
  const float* trans   = (const float*)d_in[8];
  const float* startt  = (const float*)d_in[9];
  const float* stopt   = (const float*)d_in[10];
  float* out = (float*)d_out;

  char* ws = (char*)d_ws;
  u32*   WfeatPT = (u32*)(ws);                 // 256 KB
  u32*   WemitPT = (u32*)(ws + 262144);        //  64 KB
  float* esum    = (float*)(ws + 327680);      // 512 KB
  float* emgold  = (float*)(ws + 851968);      // 512 KB

  crf_prep<<<256, 256, 0, stream>>>(Wf, We, WfeatPT, WemitPT);
  crf_fused<<<(BATCH * S_LEN) / 64, 256, LDS_TOTAL_DW * 4, stream>>>(
      seq, emb, bf, be, targets, WfeatPT, WemitPT, esum, emgold);
  crf_final<<<1, 256, 0, stream>>>(seq, targets, trans, startt, stopt,
                                   esum, emgold, out);
}
